// SelectedUnitsHead_8761733283964
// MI455X (gfx1250) — compile-verified, matrix operand order checked by reading, not run on hardware
//
#include <hip/hip_runtime.h>
#include <hip/hip_bf16.h>
#include <stdint.h>

// Problem constants (match reference)
#define B_     512
#define S_     512
#define E_     256
#define A_     1024
#define T_     256
#define MAXSEL 64
#define NEGV   (-1.0e9f)

typedef __attribute__((ext_vector_type(16))) __bf16 v16bf;
typedef __attribute__((ext_vector_type(8)))  float  v8f;

union Frag { v16bf v; unsigned u[8]; };

// d_out layout: units_logits [B,64,512] | units [B,64] | ae [B,1024] | sel_num [B]
static constexpr size_t L0 = 0;
static constexpr size_t U0 = (size_t)B_ * MAXSEL * S_;
static constexpr size_t A0 = U0 + (size_t)B_ * MAXSEL;
static constexpr size_t SN0 = A0 + (size_t)B_ * A_;

static __device__ __forceinline__ unsigned short f2b(float f) {
    return __builtin_bit_cast(unsigned short, (__bf16)f);
}
static __device__ __forceinline__ float b2f(unsigned short u) {
    return (float)__builtin_bit_cast(__bf16, u);
}
static __device__ __forceinline__ unsigned packbf(float lo, float hi) {
    return (unsigned)f2b(lo) | ((unsigned)f2b(hi) << 16);
}
// K-pair index for 16-bit A/B fragment dword j (0..7), per CDNA5 ISA
// 16-bit 16x32 layout: lanes 0-15 hold kpairs {0..3, 8..11}, lanes 16-31 {4..7, 12..15}
static __device__ __forceinline__ int kpj(int j, int half) {
    return (j < 4 ? j : j + 4) + (half ? 4 : 0);
}
// A fragment from LDS bf16 row-major [16][kldDw*2] (lowers to 2x ds_load_b128)
static __device__ __forceinline__ v16bf ldsA(const unsigned* lds, int m, int half, int kb, int kldDw) {
    Frag f;
#pragma unroll
    for (int j = 0; j < 8; ++j) f.u[j] = lds[m * kldDw + (kb >> 1) + kpj(j, half)];
    return f.v;
}
// B fragment from fragment-major swizzled panel: each lane's 8 dwords are contiguous
// -> lowers to 2x global_load_b128 instead of 8x global_load_b32
static __device__ __forceinline__ v16bf gBswz(const unsigned* panel, int nt, int kt, int Ntiles, int lane) {
    Frag f;
    const unsigned* p = panel + (((size_t)(kt * Ntiles + nt) * 32 + lane) << 3);
#pragma unroll
    for (int j = 0; j < 8; ++j) f.u[j] = p[j];
    return f.v;
}
static __device__ __forceinline__ v8f wmma_bf16(v16bf a, v16bf b, v8f c) {
    return __builtin_amdgcn_wmma_f32_16x16x32_bf16(false, a, false, b, (short)0, c, false, false);
}
static __device__ __forceinline__ float u01(unsigned x) {
    x ^= x >> 17; x *= 0xed5ad4bbu; x ^= x >> 11; x *= 0xac4c1b51u;
    x ^= x >> 15; x *= 0x31848babu; x ^= x >> 14;
    return (x >> 8) * (0.9999998f / 16777216.0f) + 1e-7f;   // in (0,1)
}

// ---- pack fp32 [N,K] weight into fragment-major bf16 WMMA B-panel.
// Layout: [ktile][ntile][lane 0..31][dword 0..7]; dword j of lane holds
// {bf16(W[n, 2kp]), bf16(W[n, 2kp+1])} with n = ntile*16 + (lane&15),
// kp = ktile*16 + kpj(j, lane>>4) -- exactly the per-lane WMMA B operand order.
__global__ void pack_frag(const float* __restrict__ src, unsigned* __restrict__ dst, int K, int N) {
    int idx = blockIdx.x * 256 + threadIdx.x;
    int total = (K / 2) * N;
    if (idx >= total) return;
    int f = idx >> 8, w = idx & 255;
    int lane = w >> 3, j = w & 7;
    int Ntiles = N >> 4;
    int kt = f / Ntiles, nt = f - kt * Ntiles;
    int m = lane & 15, half = lane >> 4;
    int kp = kt * 16 + kpj(j, half);
    int n = nt * 16 + m;
    dst[idx] = packbf(src[(size_t)n * K + 2 * kp], src[(size_t)n * K + 2 * kp + 1]);
}

// ---- the_func_embed = relu(uoh @ Wf.T + bf), [512,256]x[256,256], WMMA bf16
__global__ void tfe_kernel(const float* __restrict__ uoh, const unsigned* __restrict__ WfP,
                           const float* __restrict__ bfv, float* __restrict__ tfe) {
    __shared__ unsigned sA[16 * 128];  // 16 rows x 256 bf16
    const int lane = threadIdx.x;
    const int g0 = blockIdx.x * 16;
    for (int t = lane; t < 16 * 128; t += 32) {
        int row = t >> 7, kp = t & 127;
        const float* src = uoh + (size_t)(g0 + row) * 256 + kp * 2;
        sA[t] = packbf(src[0], src[1]);
    }
    __syncthreads();
    const int m = lane & 15, half = lane >> 4;
    for (int nt = 0; nt < 16; ++nt) {
        v8f acc = {};
        for (int kb = 0; kb < 256; kb += 32)
            acc = wmma_bf16(ldsA(sA, m, half, kb, 128), gBswz(WfP, nt, kb >> 5, 16, lane), acc);
        int n = nt * 16 + m;
#pragma unroll
        for (int r = 0; r < 8; ++r) {
            int M = r + (half ? 8 : 0);
            float v = acc[r] + bfv[n];
            tfe[(size_t)(g0 + M) * 256 + n] = v > 0.f ? v : 0.f;
        }
    }
}

// ---- key_ = ee @ Wc.T + bc with masking; store bf16 (L2-resident 16 MB)
__global__ void keys_kernel(const float* __restrict__ ee, const unsigned* __restrict__ WcP,
                            const float* __restrict__ bc, const int* __restrict__ en,
                            const float* __restrict__ nv, unsigned short* __restrict__ keysb) {
    __shared__ unsigned sA[16 * 128];  // 16 rows x 256 bf16
    const int lane = threadIdx.x;
    const int g0 = blockIdx.x * 16;    // flattened (b,s) row
    for (int t = lane; t < 16 * 128; t += 32) {
        int row = t >> 7, kp = t & 127;
        const float* src = ee + (size_t)(g0 + row) * 256 + kp * 2;
        sA[t] = packbf(src[0], src[1]);
    }
    __syncthreads();
    const int m = lane & 15, half = lane >> 4;
    for (int nt = 0; nt < 2; ++nt) {
        v8f acc = {};
        for (int kb = 0; kb < 256; kb += 32)
            acc = wmma_bf16(ldsA(sA, m, half, kb, 128), gBswz(WcP, nt, kb >> 5, 2, lane), acc);
        int n = nt * 16 + m;
#pragma unroll
        for (int r = 0; r < 8; ++r) {
            int M = r + (half ? 8 : 0);
            int g = g0 + M;
            int b = g >> 9, s = g & 511;
            float v = acc[r] + bc[n];
            if (s == S_ - 1) v = 0.f;
            if (s == en[b]) v = nv[n];
            keysb[(size_t)g * 32 + n] = f2b(v);
        }
    }
}

// ---- key_avg[b,h] = sum_{s<=en} key / en
__global__ void keyavg_kernel(const unsigned short* __restrict__ keysb,
                              const int* __restrict__ en, float* __restrict__ kavg) {
    int b = blockIdx.x, h = threadIdx.x;
    int e = en[b];
    float s = 0.f;
    for (int i = 0; i <= e; ++i) s += b2f(keysb[((size_t)b * 512 + i) * 32 + h]);
    kavg[b * 32 + h] = s / (float)e;
}

// ---- persistent 64-step autoregressive loop: 32 blocks x 16 batch rows, 128 threads (4 waves)
__global__ __launch_bounds__(128) void loop_kernel(
    const float* __restrict__ aein, const int* __restrict__ en,
    const unsigned char* __restrict__ selmask,
    const float* __restrict__ b1, const float* __restrict__ b2,
    const float* __restrict__ Wih, const float* __restrict__ Whh,
    const float* __restrict__ bih, const float* __restrict__ bhh,
    const float* __restrict__ Wp, const float* __restrict__ bp,
    const unsigned* __restrict__ W1P, const unsigned* __restrict__ W2P,
    const float* __restrict__ tfe, const unsigned short* __restrict__ keysb,
    const float* __restrict__ kavg,
    float* __restrict__ aeF, float* __restrict__ out) {
    __shared__ unsigned aeb[16 * 512];   // ae bf16 [16][1024]
    __shared__ unsigned xr[16 * 128];    // relu(x+tfe) bf16 [16][256]
    __shared__ float xg[16 * 32];
    __shared__ float gts[16 * 128];
    __shared__ float hS[16 * 32];
    __shared__ float cS[16 * 32];
    __shared__ unsigned maskS[16 * 16];  // 512-bit mask per row
    __shared__ float outS[16 * 32];
    __shared__ float kavgS[16 * 32];
    __shared__ float redV[128];
    __shared__ int redI[128];
    __shared__ int eidS[16], isendS[16], selS[16], enS[16];

    const int tid = threadIdx.x;
    const int lane = tid & 31, wave = tid >> 5;
    const int m = lane & 15, half = lane >> 4;
    const int r0 = blockIdx.x * 16;

    if (tid < 16) {
        enS[tid] = en[r0 + tid];
        eidS[tid] = 0; isendS[tid] = 0; selS[tid] = MAXSEL;
    }
    for (int t = tid; t < 512; t += 128) {
        hS[t] = 0.f; cS[t] = 0.f;
        kavgS[t] = kavg[(r0 + (t >> 5)) * 32 + (t & 31)];
    }
    for (int t = tid; t < 16 * 512; t += 128) {   // aeF was pre-filled with input ae
        int row = t >> 9, d = t & 511;
        const float* src = aeF + (size_t)(r0 + row) * 1024 + d * 2;
        aeb[t] = packbf(src[0], src[1]);
    }
    __syncthreads();
    for (int t = tid; t < 256; t += 128) {        // mask0 & ~end_pos  ==  s < en
        int row = t >> 4, w = t & 15;
        int e = enS[row], lo = w * 32;
        unsigned mv = (e <= lo) ? 0u : (e >= lo + 32 ? 0xffffffffu : ((1u << (e - lo)) - 1u));
        maskS[t] = mv;
    }
    __syncthreads();

    for (int step = 0; step < MAXSEL; ++step) {
        if (step == 1 && tid < 16) {              // re-enable EOS at step 1
            int e = enS[tid];
            maskS[tid * 16 + (e >> 5)] |= (1u << (e & 31));
        }
        __syncthreads();

        // GEMM1: x = ae @ W1.T  [16,1024]x[1024,256], 4 waves x 4 N-tiles
        for (int nt = wave * 4; nt < wave * 4 + 4; ++nt) {
            v8f acc = {};
            for (int kb = 0; kb < 1024; kb += 32) {
                int kt = kb >> 5;
                if (kt + 1 < 32)   // stream next fragment of L2-resident W1 panel
                    __builtin_prefetch(W1P + (((size_t)((kt + 1) * 16 + nt) * 32 + lane) << 3), 0, 1);
                acc = wmma_bf16(ldsA(aeb, m, half, kb, 512), gBswz(W1P, nt, kt, 16, lane), acc);
            }
            int n = nt * 16 + m;
#pragma unroll
            for (int rr = 0; rr < 8; ++rr) {
                int M = rr + (half ? 8 : 0);
                float v = acc[rr] + b1[n] + tfe[(size_t)(r0 + M) * 256 + n];
                v = v > 0.f ? v : 0.f;                       // relu(x + the_func_embed)
                ((unsigned short*)xr)[M * 256 + n] = f2b(v);
            }
        }
        __syncthreads();

        // GEMM2: [16,256]x[256,32], waves 0/1
        if (wave < 2) {
            int nt = wave;
            v8f acc = {};
            for (int kb = 0; kb < 256; kb += 32)
                acc = wmma_bf16(ldsA(xr, m, half, kb, 128), gBswz(W2P, nt, kb >> 5, 2, lane), acc);
            int n = nt * 16 + m;
#pragma unroll
            for (int rr = 0; rr < 8; ++rr) {
                int M = rr + (half ? 8 : 0);
                xg[M * 32 + n] = acc[rr] + b2[n];
            }
        }
        __syncthreads();

        // LSTM gates
        for (int t = tid; t < 16 * 128; t += 128) {
            int row = t >> 7, j = t & 127;
            float a = bih[j] + bhh[j];
            const float* wi = Wih + j * 32;
            const float* wh = Whh + j * 32;
#pragma unroll 8
            for (int k = 0; k < 32; ++k)
                a += xg[row * 32 + k] * wi[k] + hS[row * 32 + k] * wh[k];
            gts[t] = a;
        }
        __syncthreads();
        for (int t = tid; t < 512; t += 128) {
            int row = t >> 5, k = t & 31;
            float gi = gts[row * 128 + k];
            float gf = gts[row * 128 + 32 + k];
            float gg = gts[row * 128 + 64 + k];
            float go = gts[row * 128 + 96 + k];
            gi = 1.f / (1.f + __expf(-gi));
            gf = 1.f / (1.f + __expf(-gf));
            go = 1.f / (1.f + __expf(-go));
            float c = gf * cS[t] + gi * tanhf(gg);
            cS[t] = c;
            hS[t] = go * tanhf(c);
        }
        __syncthreads();

        // logits y = h . key_[b,s,:], masked; write logits; Gumbel-max sample (deterministic hash)
        {
            int sub = tid & 7, row = tid >> 3;
            int g = r0 + row;
            float best = -3.4e38f; int bi = 0;
            const float* hrow = hS + row * 32;
            for (int s = sub * 64; s < sub * 64 + 64; ++s) {
                const unsigned short* kr = keysb + ((size_t)g * 512 + s) * 32;
                float y = 0.f;
#pragma unroll 8
                for (int h = 0; h < 32; ++h) y += b2f(kr[h]) * hrow[h];
                bool mk = (maskS[row * 16 + (s >> 5)] >> (s & 31)) & 1u;
                float lg = mk ? y : NEGV;
                out[L0 + ((size_t)g * MAXSEL + step) * 512 + s] = lg;
                unsigned hx = (unsigned)g * 2654435761u ^ (unsigned)step * 40503u ^ (unsigned)s * 2246822519u;
                float sc = lg - __logf(-__logf(u01(hx)));
                if (sc > best) { best = sc; bi = s; }
            }
            redV[tid] = best; redI[tid] = bi;
        }
        __syncthreads();
        if ((tid & 7) == 0) {
            int row = tid >> 3;
            float best = redV[tid]; int bi = redI[tid];
            for (int q = 1; q < 8; ++q)
                if (redV[tid + q] > best) { best = redV[tid + q]; bi = redI[tid + q]; }
            eidS[row] = bi;
            maskS[row * 16 + (bi >> 5)] &= ~(1u << (bi & 31));
            if (bi == enS[row]) { isendS[row] = 1; selS[row] = step; }
            out[U0 + (size_t)(r0 + row) * MAXSEL + step] = (float)bi;
        }
        __syncthreads();
        for (int t = tid; t < 512; t += 128) {
            int row = t >> 5, h = t & 31;
            int g = r0 + row;
            outS[t] = b2f(keysb[((size_t)g * 512 + eidS[row]) * 32 + h]) - kavgS[t];
        }
        __syncthreads();

        // ae += (out @ Wp.T + bp) * ~is_end ; refresh bf16 copy
        for (int t = tid; t < 16 * 1024; t += 128) {
            int row = t >> 10, a = t & 1023;
            if (!isendS[row]) {
                int g = r0 + row;
                float d = bp[a];
                const float* wp = Wp + a * 32;
                const float* o = outS + row * 32;
#pragma unroll 8
                for (int h = 0; h < 32; ++h) d += o[h] * wp[h];
                float v = aeF[(size_t)g * 1024 + a] + d;
                aeF[(size_t)g * 1024 + a] = v;
                ((unsigned short*)aeb)[row * 1024 + a] = f2b(v);
            }
        }
        __syncthreads();
    }

    // finalize: sel_num, ae, no-select patches
    for (int t = tid; t < 16; t += 128) {
        int g = r0 + t;
        out[SN0 + g] = (float)((selmask[g] != 0) ? selS[t] : 0);
    }
    for (int t = tid; t < 16 * 1024; t += 128) {
        int row = t >> 10, a = t & 1023;
        int g = r0 + row;
        out[A0 + (size_t)g * 1024 + a] =
            (selmask[g] != 0) ? aeF[(size_t)g * 1024 + a] : aein[(size_t)g * 1024 + a];
    }
    for (int row = 0; row < 16; ++row) {
        int g = r0 + row;
        if (selmask[g] == 0) {
            for (int t = tid; t < MAXSEL * 512; t += 128)
                out[L0 + (size_t)g * MAXSEL * 512 + t] = NEGV;
            for (int t = tid; t < MAXSEL; t += 128)
                out[U0 + (size_t)g * MAXSEL + t] = (float)(S_ - 1);
        }
    }
}

extern "C" void kernel_launch(void* const* d_in, const int* in_sizes, int n_in,
                              void* d_out, int out_size, void* d_ws, size_t ws_size,
                              hipStream_t stream) {
    (void)in_sizes; (void)n_in; (void)out_size; (void)ws_size;
    const float* ae  = (const float*)d_in[0];
    const float* ee  = (const float*)d_in[1];
    const float* uoh = (const float*)d_in[2];
    const int* en    = (const int*)d_in[3];
    const unsigned char* sm = (const unsigned char*)d_in[4];
    const float* Wf  = (const float*)d_in[5];
    const float* bfv = (const float*)d_in[6];
    const float* Wc  = (const float*)d_in[7];
    const float* bc  = (const float*)d_in[8];
    const float* W1  = (const float*)d_in[9];
    const float* b1  = (const float*)d_in[10];
    const float* W2  = (const float*)d_in[11];
    const float* b2  = (const float*)d_in[12];
    const float* Wih = (const float*)d_in[13];
    const float* Whh = (const float*)d_in[14];
    const float* bih = (const float*)d_in[15];
    const float* bhh = (const float*)d_in[16];
    const float* Wp  = (const float*)d_in[17];
    const float* bp  = (const float*)d_in[18];
    const float* nv  = (const float*)d_in[19];

    char* ws = (char*)d_ws;
    unsigned short* keysb = (unsigned short*)(ws + 0);         // 16 MB bf16 keys
    float* kavg = (float*)(ws + 16777216);                     // 64 KB
    float* tfe  = (float*)(ws + 16842752);                     // 512 KB
    float* aeF  = (float*)(ws + 17367040);                     // 2 MB running ae (f32)
    unsigned* WcP = (unsigned*)(ws + 19464192);                // 16 KB
    unsigned* WfP = (unsigned*)(ws + 19480576);                // 128 KB
    unsigned* W1P = (unsigned*)(ws + 19611648);                // 512 KB
    unsigned* W2P = (unsigned*)(ws + 20135936);                // 16 KB
    float* out = (float*)d_out;

    hipMemcpyAsync(aeF, ae, (size_t)B_ * A_ * sizeof(float), hipMemcpyDeviceToDevice, stream);

    pack_frag<<<dim3((128 * 32 + 255) / 256), 256, 0, stream>>>(Wc, WcP, 256, 32);
    pack_frag<<<dim3((128 * 256 + 255) / 256), 256, 0, stream>>>(Wf, WfP, 256, 256);
    pack_frag<<<dim3((512 * 256 + 255) / 256), 256, 0, stream>>>(W1, W1P, 1024, 256);
    pack_frag<<<dim3((128 * 32 + 255) / 256), 256, 0, stream>>>(W2, W2P, 256, 32);

    tfe_kernel<<<B_ / 16, 32, 0, stream>>>(uoh, WfP, bfv, tfe);
    keys_kernel<<<(B_ * S_) / 16, 32, 0, stream>>>(ee, WcP, bc, en, nv, keysb);
    keyavg_kernel<<<B_, 32, 0, stream>>>(keysb, en, kavg);
    loop_kernel<<<B_ / 16, 128, 0, stream>>>(ae, en, sm, b1, b2, Wih, Whh, bih, bhh, Wp, bp,
                                             W1P, W2P, tfe, keysb, kavg, aeF, out);
}
